// CalibrationModelObsGridGeometry_20555713478969
// MI455X (gfx1250) — matrix operand order repaired and checked by reading, not compiled, not run on hardware
//
#include <hip/hip_runtime.h>

#define IMGS 16
#define Hh 512
#define Ww 240
#define MINSZ 300
#define NSIG 10
#define KS 481
#define PADH 240
#define NF 22
#define HID 64

typedef __attribute__((ext_vector_type(16))) __bf16 v16bf;
typedef __attribute__((ext_vector_type(8)))  float  v8f;
typedef __attribute__((ext_vector_type(4)))  float  f32x4;

struct Frag32B { f32x4 lo, hi; };

static __device__ __forceinline__ v16bf frag_ld2(const __bf16* p0, const __bf16* p1) {
  Frag32B s;
  s.lo = *(const f32x4*)p0;
  s.hi = *(const f32x4*)p1;
  return __builtin_bit_cast(v16bf, s);
}

// Issue a per-lane async global->LDS 16-byte copy (ASYNCcnt-tracked).
static __device__ __forceinline__ void async_g2l_b128(unsigned int lds_off,
                                                      unsigned long long gaddr) {
  asm volatile("global_load_async_to_lds_b128 %0, %1, off"
               :: "v"(lds_off), "v"(gaddr) : "memory");
}
static __device__ __forceinline__ void wait_asynccnt0() {
  asm volatile("s_wait_asynccnt 0" ::: "memory");
}

// ---------------------------------------------------------------- mask kernel
__global__ __launch_bounds__(256) void mask_kernel(const float* __restrict__ fy,
                                                   int* __restrict__ msk) {
  int img = blockIdx.x;
  const float* p = fy + (size_t)img * Hh * Ww;
  int ok = 1;
  for (int i = threadIdx.x; i < MINSZ * Ww; i += 256) {
    float v = p[i];
    ok &= __builtin_isfinite(v) ? 1 : 0;
  }
  __shared__ int red[256];
  red[threadIdx.x] = ok;
  __syncthreads();
  for (int s = 128; s > 0; s >>= 1) {
    if (threadIdx.x < s) red[threadIdx.x] &= red[threadIdx.x + s];
    __syncthreads();
  }
  if (threadIdx.x == 0) msk[img] = red[0];
}

// ------------------------------------------------------------- weight prep
// w1b: [64][9][32] bf16 (channels >=22 zero)   w2b: [64][9][64] bf16
__global__ __launch_bounds__(256) void wprep_kernel(const float* __restrict__ w1,
                                                    const float* __restrict__ w2,
                                                    __bf16* __restrict__ w1b,
                                                    __bf16* __restrict__ w2b) {
  int i = blockIdx.x * 256 + threadIdx.x;
  if (i < HID * 9 * 32) {
    int c = i & 31, t = (i >> 5) % 9, o = i / (9 * 32);
    float v = (c < NF) ? w1[((o * NF + c) * 3 + t / 3) * 3 + (t % 3)] : 0.0f;
    w1b[i] = (__bf16)v;
  }
  int j = i - HID * 9 * 32;
  if (j >= 0 && j < HID * 9 * 64) {
    int c = j & 63, t = (j >> 6) % 9, o = j / (9 * 64);
    w2b[j] = (__bf16)w2[((o * 64 + c) * 3 + t / 3) * 3 + (t % 3)];
  }
}

// ------------------------------------------------------------ feature kernel
// Computes cal_input = -concat(feats(fy), feats(fs)), stores bf16 NHWC (C=32,
// channels 22..31 zero), and fixed-order per-block BN partial sums.
__global__ __launch_bounds__(256) void feature_kernel(
    const float* __restrict__ sv_uncal, const float* __restrict__ sv_bg,
    const float* __restrict__ gk, const int* __restrict__ msk,
    __bf16* __restrict__ feats, float* __restrict__ partials) {
  __shared__ float colY[8][513];
  __shared__ float colS[8][513];
  __shared__ float kern[NSIG * KS];
  __shared__ float wred[8][44];

  int img = blockIdx.y;
  int x0 = blockIdx.x * 8;
  int tid = threadIdx.x;

  for (int i = tid; i < NSIG * KS; i += 256) kern[i] = gk[i];

  const float* py = sv_uncal + (size_t)img * Hh * Ww;
  const float* ps = sv_bg + (size_t)img * Hh * Ww;
  for (int i = tid; i < 8 * Hh; i += 256) {
    int c = i & 7, r = i >> 3;
    float vy = py[r * Ww + x0 + c];
    colY[c][r] = (vy == vy) ? vy : 0.0f;   // nan_to_num (inputs have NaN only)
    colS[c][r] = ps[r * Ww + x0 + c];
  }
  __syncthreads();

  int valid = msk[img];
  int c = tid & 7, rr = tid >> 3;  // 8 cols x 32 row-phases
  float sum[22], sq[22];
#pragma unroll
  for (int k = 0; k < 22; ++k) { sum[k] = 0.f; sq[k] = 0.f; }

  for (int chunk = 0; chunk < 16; ++chunk) {
    int r = rr + chunk * 32;
    float g0[NSIG], g1[NSIG];
#pragma unroll
    for (int s = 0; s < NSIG; ++s) { g0[s] = 0.f; g1[s] = 0.f; }
    for (int t = 0; t < KS; ++t) {
      int j = r + t - PADH;
      j = j < 0 ? -j : j;
      j = j > (Hh - 1) ? 2 * (Hh - 1) - j : j;
      float vy = colY[c][j];
      float vs = colS[c][j];
#pragma unroll
      for (int s = 0; s < NSIG; ++s) {
        float kv = kern[s * KS + t];
        g0[s] += vy * kv;
        g1[s] += vs * kv;
      }
    }
    float fxY = py[r * Ww + x0 + c];       // raw (possibly NaN) value
    float fxS = colS[c][r];
    float f[22];
    f[0] = -(g0[0] - fxY);
#pragma unroll
    for (int s = 1; s < NSIG; ++s) f[s] = -(g0[s] - g0[s - 1]);
    f[10] = -g0[NSIG - 1];
    f[11] = -(g1[0] - fxS);
#pragma unroll
    for (int s = 1; s < NSIG; ++s) f[11 + s] = -(g1[s] - g1[s - 1]);
    f[21] = -g1[NSIG - 1];

    __align__(16) __bf16 ob[32];
#pragma unroll
    for (int k = 0; k < 22; ++k) ob[k] = (__bf16)f[k];
#pragma unroll
    for (int k = 22; k < 32; ++k) ob[k] = (__bf16)0.0f;
    f32x4* dst = (f32x4*)(feats + ((size_t)(img * Hh + r) * Ww + x0 + c) * 32);
#pragma unroll
    for (int q = 0; q < 4; ++q) dst[q] = ((f32x4*)ob)[q];

    if (valid) {
#pragma unroll
      for (int k = 0; k < 22; ++k) { sum[k] += f[k]; sq[k] += f[k] * f[k]; }
    }
  }

  int wv = tid >> 5, lane = tid & 31;
#pragma unroll
  for (int k = 0; k < 22; ++k) {
    float a = sum[k], b = sq[k];
    for (int o = 16; o > 0; o >>= 1) { a += __shfl_down(a, o); b += __shfl_down(b, o); }
    if (lane == 0) { wred[wv][k] = a; wred[wv][22 + k] = b; }
  }
  __syncthreads();
  if (tid < 44) {
    float acc = 0.f;
    for (int w = 0; w < 8; ++w) acc += wred[w][tid];
    partials[(size_t)(blockIdx.y * 30 + blockIdx.x) * 44 + tid] = acc;
  }
}

// ------------------------------------------------------------- BN finalize
// normP: [0..31]=mean, [32..63]=scale, [64..95]=beta (padded channels: 0)
__global__ void bn_finalize_kernel(const float* __restrict__ partials,
                                   const int* __restrict__ msk,
                                   const float* __restrict__ gamma,
                                   const float* __restrict__ beta,
                                   float* __restrict__ normP) {
  int tid = threadIdx.x;
  int cnt = 0;
  for (int i = 0; i < IMGS; ++i) cnt += msk[i];
  float n = (float)cnt * (float)(Hh * Ww);
  if (tid < 32) {
    float mean = 0.f, scale = 0.f, bet = 0.f;
    if (tid < NF) {
      float s = 0.f, q = 0.f;
      for (int b = 0; b < 480; ++b) {
        s += partials[(size_t)b * 44 + tid];
        q += partials[(size_t)b * 44 + 22 + tid];
      }
      mean = s / n;
      float var = q / n - mean * mean;
      scale = gamma[tid] * rsqrtf(var + 1e-5f);
      bet = beta[tid];
    }
    normP[tid] = mean;
    normP[32 + tid] = scale;
    normP[64 + tid] = bet;
  }
}

// ---------------------------------------------------- conv1 (32pad -> 64) WMMA
// Staging keeps the VGPR path: it applies the BN affine transform in flight.
__global__ __launch_bounds__(256) void conv1_kernel(
    const __bf16* __restrict__ feats, const __bf16* __restrict__ w1b,
    const float* __restrict__ bias, const int* __restrict__ msk,
    const float* __restrict__ normP, __bf16* __restrict__ h1) {
  int img = blockIdx.z;
  if (!msk[img]) return;
  int y = blockIdx.y;
  int x0 = blockIdx.x * 64;
  int tid = threadIdx.x;
  __shared__ __align__(16) __bf16 act[3 * 66 * 32];

  const size_t inBase = (size_t)img * Hh * Ww * 32;
  const size_t outBase = (size_t)img * Hh * Ww * 64;
  for (int i = tid; i < 3 * 66 * 4; i += 256) {
    int q = i & 3, jc = (i >> 2) % 66, rr = i / (66 * 4);
    int yy = y + rr - 1, xx = x0 + jc - 1;
    f32x4 v = {0.f, 0.f, 0.f, 0.f};
    if (yy >= 0 && yy < Hh && xx >= 0 && xx < Ww) {
      f32x4 raw = *(const f32x4*)(feats + inBase + ((size_t)yy * Ww + xx) * 32 + q * 8);
      __bf16* rb = (__bf16*)&raw;
      __align__(16) __bf16 nb[8];
#pragma unroll
      for (int e = 0; e < 8; ++e) {
        int cc = q * 8 + e;
        float xv = (float)rb[e];
        float xn = (xv - normP[cc]) * normP[32 + cc] + normP[64 + cc];
        nb[e] = (__bf16)xn;
      }
      v = *(f32x4*)nb;
    }
    ((f32x4*)act)[i] = v;
  }
  __syncthreads();

  int wv = tid >> 5, lane = tid & 31;
  int lm = lane & 15, half = lane >> 4;
  for (int ti = wv; ti < 16; ti += 8) {
    int m0 = (ti >> 2) * 16, p0 = (ti & 3) * 16;
    v8f acc = {0.f, 0.f, 0.f, 0.f, 0.f, 0.f, 0.f, 0.f};
#pragma unroll
    for (int t = 0; t < 9; ++t) {
      int dy = t / 3, dx = t % 3;
      const __bf16* brow = act + ((size_t)(dy * 66 + (p0 + lm + dx))) * 32;
      const __bf16* wb = w1b + ((size_t)(m0 + lm) * 9 + t) * 32;
      v16bf A = frag_ld2(wb + 16 * half, wb + 16 * half + 8);
      v16bf B = frag_ld2(brow + 16 * half, brow + 16 * half + 8);
      acc = __builtin_amdgcn_wmma_f32_16x16x32_bf16(false, A, false, B,
                                                    (short)0, acc, false, false);
    }
    int x = x0 + p0 + lm;
    if (x < Ww) {
      __align__(16) __bf16 ob[8];
      int mb = m0 + 8 * half;
#pragma unroll
      for (int r = 0; r < 8; ++r) {
        float v = acc[r] + bias[mb + r];
        ob[r] = (__bf16)(v > 0.f ? v : 0.f);
      }
      *(f32x4*)(h1 + outBase + ((size_t)y * Ww + x) * 64 + mb) = *(f32x4*)ob;
    }
  }
}

// ------------------------------------------------------ conv2 (64 -> 64) WMMA
// Activation halo tile staged with ASYNCcnt-tracked global->LDS async copies;
// out-of-range (SAME padding) lanes zero-fill LDS via DS stores (disjoint).
__global__ __launch_bounds__(256) void conv2_kernel(
    const __bf16* __restrict__ h1, const __bf16* __restrict__ w2b,
    const float* __restrict__ bias, const int* __restrict__ msk,
    __bf16* __restrict__ h2) {
  int img = blockIdx.z;
  if (!msk[img]) return;
  int y = blockIdx.y;
  int x0 = blockIdx.x * 64;
  int tid = threadIdx.x;
  __shared__ __align__(16) __bf16 act[3 * 66 * 64];

  const size_t imgBase = (size_t)img * Hh * Ww * 64;
  const unsigned int actBase = (unsigned int)(uintptr_t)(void*)act;
  for (int i = tid; i < 3 * 66 * 8; i += 256) {
    int q = i & 7, jc = (i >> 3) % 66, rr = i / (66 * 8);
    int yy = y + rr - 1, xx = x0 + jc - 1;
    if (yy >= 0 && yy < Hh && xx >= 0 && xx < Ww) {
      unsigned long long ga = (unsigned long long)(uintptr_t)
          (h1 + imgBase + ((size_t)yy * Ww + xx) * 64 + q * 8);
      async_g2l_b128(actBase + (unsigned int)i * 16u, ga);
    } else {
      f32x4 z = {0.f, 0.f, 0.f, 0.f};
      ((f32x4*)act)[i] = z;
    }
  }
  wait_asynccnt0();
  __syncthreads();

  int wv = tid >> 5, lane = tid & 31;
  int lm = lane & 15, half = lane >> 4;
  for (int ti = wv; ti < 16; ti += 8) {
    int m0 = (ti >> 2) * 16, p0 = (ti & 3) * 16;
    v8f acc = {0.f, 0.f, 0.f, 0.f, 0.f, 0.f, 0.f, 0.f};
#pragma unroll
    for (int t = 0; t < 9; ++t) {
      int dy = t / 3, dx = t % 3;
      const __bf16* brow = act + ((size_t)(dy * 66 + (p0 + lm + dx))) * 64;
      const __bf16* wb = w2b + ((size_t)(m0 + lm) * 9 + t) * 64;
#pragma unroll
      for (int kc = 0; kc < 2; ++kc) {
        v16bf A = frag_ld2(wb + kc * 32 + 16 * half, wb + kc * 32 + 16 * half + 8);
        v16bf B = frag_ld2(brow + kc * 32 + 16 * half, brow + kc * 32 + 16 * half + 8);
        acc = __builtin_amdgcn_wmma_f32_16x16x32_bf16(false, A, false, B,
                                                      (short)0, acc, false, false);
      }
    }
    int x = x0 + p0 + lm;
    if (x < Ww) {
      __align__(16) __bf16 ob[8];
      int mb = m0 + 8 * half;
#pragma unroll
      for (int r = 0; r < 8; ++r) {
        float v = acc[r] + bias[mb + r];
        ob[r] = (__bf16)(v > 0.f ? v : 0.f);
      }
      *(f32x4*)(h2 + imgBase + ((size_t)y * Ww + x) * 64 + mb) = *(f32x4*)ob;
    }
  }
}

// -------------------------------------------- conv3 (64 -> 1) + fs + masking
__global__ __launch_bounds__(256) void conv3_kernel(
    const __bf16* __restrict__ h2, const float* __restrict__ w3,
    const float* __restrict__ b3, const float* __restrict__ sv_bg,
    const int* __restrict__ msk, float* __restrict__ out_cal,
    float* __restrict__ out_msk) {
  int img = blockIdx.z;
  int y0 = blockIdx.y * 4, x0 = blockIdx.x * 64;
  int tid = threadIdx.x;
  int ry = tid >> 6, cx = tid & 63;
  int y = y0 + ry, x = x0 + cx;
  size_t pix = (size_t)(img * Hh + y) * Ww + x;
  if (!msk[img]) {
    if (x < Ww) { out_cal[pix] = 0.f; out_msk[pix] = 0.f; }
    return;
  }
  __shared__ __align__(16) __bf16 act[6 * 66 * 64];
  __shared__ float wl[576];
  for (int i = tid; i < 576; i += 256) wl[i] = w3[i];
  const size_t imgBase = (size_t)img * Hh * Ww * 64;
  const unsigned int actBase = (unsigned int)(uintptr_t)(void*)act;
  for (int i = tid; i < 6 * 66 * 8; i += 256) {
    int q = i & 7, jc = (i >> 3) % 66, rr = i / (66 * 8);
    int yy = y0 + rr - 1, xx = x0 + jc - 1;
    if (yy >= 0 && yy < Hh && xx >= 0 && xx < Ww) {
      unsigned long long ga = (unsigned long long)(uintptr_t)
          (h2 + imgBase + ((size_t)yy * Ww + xx) * 64 + q * 8);
      async_g2l_b128(actBase + (unsigned int)i * 16u, ga);
    } else {
      f32x4 z = {0.f, 0.f, 0.f, 0.f};
      ((f32x4*)act)[i] = z;
    }
  }
  wait_asynccnt0();
  __syncthreads();

  float acc = b3[0];
#pragma unroll
  for (int t = 0; t < 9; ++t) {
    int dy = t / 3, dx = t % 3;
    const __bf16* arow = act + ((size_t)((ry + dy) * 66 + (cx + dx))) * 64;
#pragma unroll 8
    for (int ch = 0; ch < 64; ++ch) acc += wl[ch * 9 + t] * (float)arow[ch];
  }
  if (x < Ww) {
    out_cal[pix] = acc + sv_bg[pix];
    out_msk[pix] = 1.0f;
  }
}

// ------------------------------------------------------------------- launch
extern "C" void kernel_launch(void* const* d_in, const int* in_sizes, int n_in,
                              void* d_out, int out_size, void* d_ws, size_t ws_size,
                              hipStream_t stream) {
  (void)in_sizes; (void)n_in; (void)out_size; (void)ws_size;
  const float* sv_uncal = (const float*)d_in[0];
  const float* sv_bg    = (const float*)d_in[1];
  const float* gk       = (const float*)d_in[2];
  const float* gamma    = (const float*)d_in[3];
  const float* beta     = (const float*)d_in[4];
  const float* w1       = (const float*)d_in[5];
  const float* b1       = (const float*)d_in[6];
  const float* w2       = (const float*)d_in[7];
  const float* b2       = (const float*)d_in[8];
  const float* w3       = (const float*)d_in[9];
  const float* b3       = (const float*)d_in[10];

  char* ws = (char*)d_ws;
  int*    msk      = (int*)(ws + 0);                 // 16 ints
  float*  normP    = (float*)(ws + 256);             // 96 floats
  float*  partials = (float*)(ws + 1024);            // 480*44 floats
  __bf16* w1b      = (__bf16*)(ws + 90112);          // 64*9*32
  __bf16* w2b      = (__bf16*)(ws + 131072);         // 64*9*64
  __bf16* feats    = (__bf16*)(ws + 262144);                         // 16*512*240*32
  __bf16* h1       = (__bf16*)(ws + 262144 + 125829120ULL);          // 16*512*240*64
  __bf16* h2       = (__bf16*)(ws + 262144 + 125829120ULL + 251658240ULL);

  float* out_cal = (float*)d_out;
  float* out_msk = out_cal + (size_t)IMGS * Hh * Ww;

  mask_kernel<<<IMGS, 256, 0, stream>>>(sv_uncal, msk);
  wprep_kernel<<<216, 256, 0, stream>>>(w1, w2, w1b, w2b);
  feature_kernel<<<dim3(30, IMGS), 256, 0, stream>>>(sv_uncal, sv_bg, gk, msk, feats, partials);
  bn_finalize_kernel<<<1, 64, 0, stream>>>(partials, msk, gamma, beta, normP);
  conv1_kernel<<<dim3(4, Hh, IMGS), 256, 0, stream>>>(feats, w1b, b1, msk, normP, h1);
  conv2_kernel<<<dim3(4, Hh, IMGS), 256, 0, stream>>>(h1, w2b, b2, msk, h2);
  conv3_kernel<<<dim3(4, Hh / 4, IMGS), 256, 0, stream>>>(h2, w3, b3, sv_bg, msk, out_cal, out_msk);
}
